// CentersDistance_54975581388961
// MI455X (gfx1250) — matrix operand order repaired and checked by reading, not compiled, not run on hardware
//
#include <hip/hip_runtime.h>

typedef __attribute__((ext_vector_type(16))) __bf16 v16bf;
typedef __attribute__((ext_vector_type(8)))  float  v8f;

#define D_DIM 1024
#define N_DIM 8192
#define K_DIM 1024
#define BLK   128
#define KC    32
#define NITER (D_DIM / KC)
#define LDSS  40   // LDS row stride in bf16 elements = 80 bytes (multiple of 16B)

static __device__ __forceinline__ unsigned short f32_to_bf16_rne(float x) {
  unsigned int u = __float_as_uint(x);
  u += 0x7FFFu + ((u >> 16) & 1u);
  return (unsigned short)(u >> 16);
}
static __device__ __forceinline__ float bf16_bits_to_f32(unsigned short hbits) {
  return __uint_as_float(((unsigned int)hbits) << 16);
}
static __device__ __forceinline__ void split4(const float4 f, ushort4& hi, ushort4& lo) {
  hi.x = f32_to_bf16_rne(f.x); lo.x = f32_to_bf16_rne(f.x - bf16_bits_to_f32(hi.x));
  hi.y = f32_to_bf16_rne(f.y); lo.y = f32_to_bf16_rne(f.y - bf16_bits_to_f32(hi.y));
  hi.z = f32_to_bf16_rne(f.z); lo.z = f32_to_bf16_rne(f.z - bf16_bits_to_f32(hi.z));
  hi.w = f32_to_bf16_rne(f.w); lo.w = f32_to_bf16_rne(f.w - bf16_bits_to_f32(hi.w));
}

union FragCast { uint4 u[2]; v16bf v; };

// One wave per row: sum of squares, fp32 exact.
__global__ void __launch_bounds__(256)
rownorm_kernel(const float* __restrict__ X, float* __restrict__ out) {
  const int wave = (blockIdx.x * blockDim.x + threadIdx.x) >> 5;
  const int lane = threadIdx.x & 31;
  const float* row = X + (size_t)wave * D_DIM;
  float s = 0.0f;
  for (int c = lane; c < D_DIM; c += 32) {
    float v = row[c];
    s += v * v;
  }
#pragma unroll
  for (int off = 16; off > 0; off >>= 1) s += __shfl_xor(s, off, 32);
  if (lane == 0) out[wave] = s;
}

__global__ void __launch_bounds__(256)
cdist_wmma_kernel(const float* __restrict__ inputs,   // (N, D) fp32
                  const float* __restrict__ centers,  // (K, D) fp32
                  const float* __restrict__ xsq,      // (N,)
                  const float* __restrict__ csq,      // (K,)
                  float* __restrict__ out) {          // (K, N)
  // Double-buffered staging: 2 x (4 x 128 x 40 x 2B) = 80 KB (<320 KB/WGP)
  __shared__ unsigned short sAhi[2][BLK * LDSS];
  __shared__ unsigned short sAlo[2][BLK * LDSS];
  __shared__ unsigned short sBhi[2][BLK * LDSS];
  __shared__ unsigned short sBlo[2][BLK * LDSS];

  const int tid  = threadIdx.x;
  const int lane = tid & 31;
  const int wv   = tid >> 5;        // 8 waves per block (wave32)
  const int r    = lane & 15;       // row/col within 16-wide tile
  const int h    = lane >> 4;       // half-wave select

  const int mBase = blockIdx.y * BLK;        // centers (K) macro offset
  const int nBase = blockIdx.x * BLK;        // inputs  (N) macro offset
  const int wmOff = (wv >> 2) * 64;          // wave: 64 rows (4 M-tiles)
  const int wnOff = (wv & 3) * 32;           // wave: 32 cols (2 N-tiles)

  const v8f vzero = {0.f, 0.f, 0.f, 0.f, 0.f, 0.f, 0.f, 0.f};
  v8f acc[4][2];
#pragma unroll
  for (int mt = 0; mt < 4; ++mt)
#pragma unroll
    for (int nt = 0; nt < 2; ++nt) acc[mt][nt] = vzero;

  // Staging assignment: thread -> (row0 = tid/8 + 32*i, col = (tid%8)*4 floats)
  const int sRow0 = tid >> 3;
  const int sCol  = (tid & 7) << 2;
  const float* gA = centers + (size_t)(mBase + sRow0) * D_DIM + sCol;
  const float* gB = inputs  + (size_t)(nBase + sRow0) * D_DIM + sCol;

  // Prologue: prefetch chunk 0 into registers.
  float4 pfA[4], pfB[4];
#pragma unroll
  for (int i = 0; i < 4; ++i) {
    pfA[i] = *(const float4*)(gA + (size_t)i * 32 * D_DIM);
    pfB[i] = *(const float4*)(gB + (size_t)i * 32 * D_DIM);
  }

  int p = 1;
  for (int it = 0; it < NITER; ++it) {
    p ^= 1;

    // ---- Convert the register-held chunk (it) into LDS buffer p.
#pragma unroll
    for (int i = 0; i < 4; ++i) {
      const int row = sRow0 + 32 * i;
      ushort4 hi, lo;
      split4(pfA[i], hi, lo);
      *(ushort4*)(&sAhi[p][row * LDSS + sCol]) = hi;
      *(ushort4*)(&sAlo[p][row * LDSS + sCol]) = lo;
      split4(pfB[i], hi, lo);
      *(ushort4*)(&sBhi[p][row * LDSS + sCol]) = hi;
      *(ushort4*)(&sBlo[p][row * LDSS + sCol]) = lo;
    }
    __syncthreads();  // single barrier per iteration (ping-pong buffers)

    // ---- Issue global loads for chunk it+1; latency hides under the WMMAs.
    if (it + 1 < NITER) {
      const size_t dNext = (size_t)(it + 1) * KC;
#pragma unroll
      for (int i = 0; i < 4; ++i) {
        pfA[i] = *(const float4*)(gA + dNext + (size_t)i * 32 * D_DIM);
        pfB[i] = *(const float4*)(gB + dNext + (size_t)i * 32 * D_DIM);
      }
    }

    // ---- Load fragments per gfx1250 WMMA VGPR layouts.
    // A 16x32 bf16: lane(r,h) elems[0:8]=K[8h..8h+8), elems[8:16]=K[16+8h..16+8h+8)
    // B 32x16 bf16: lane(r,h) holds column r, elems[0:8]=K[8h..8h+8), elems[8:16]=K[16+8h..16+8h+8)
    v16bf aHi[4], aLo[4], bHi[2], bLo[2];
#pragma unroll
    for (int mt = 0; mt < 4; ++mt) {
      const int arow = wmOff + mt * 16 + r;
      FragCast f0, f1;
      f0.u[0] = *(const uint4*)(&sAhi[p][arow * LDSS + 8 * h]);
      f0.u[1] = *(const uint4*)(&sAhi[p][arow * LDSS + 16 + 8 * h]);
      aHi[mt] = f0.v;
      f1.u[0] = *(const uint4*)(&sAlo[p][arow * LDSS + 8 * h]);
      f1.u[1] = *(const uint4*)(&sAlo[p][arow * LDSS + 16 + 8 * h]);
      aLo[mt] = f1.v;
    }
#pragma unroll
    for (int nt = 0; nt < 2; ++nt) {
      const int bcol = wnOff + nt * 16 + r;
      FragCast f0, f1;
      f0.u[0] = *(const uint4*)(&sBhi[p][bcol * LDSS + 8 * h]);
      f0.u[1] = *(const uint4*)(&sBhi[p][bcol * LDSS + 16 + 8 * h]);
      bHi[nt] = f0.v;
      f1.u[0] = *(const uint4*)(&sBlo[p][bcol * LDSS + 8 * h]);
      f1.u[1] = *(const uint4*)(&sBlo[p][bcol * LDSS + 16 + 8 * h]);
      bLo[nt] = f1.v;
    }

    // ---- bf16x3 accumulate: hi*hi + hi*lo + lo*hi (fp32 accumulators)
#pragma unroll
    for (int mt = 0; mt < 4; ++mt)
#pragma unroll
      for (int nt = 0; nt < 2; ++nt) {
        acc[mt][nt] = __builtin_amdgcn_wmma_f32_16x16x32_bf16(
            false, aHi[mt], false, bHi[nt], (short)0, acc[mt][nt], false, false);
        acc[mt][nt] = __builtin_amdgcn_wmma_f32_16x16x32_bf16(
            false, aHi[mt], false, bLo[nt], (short)0, acc[mt][nt], false, false);
        acc[mt][nt] = __builtin_amdgcn_wmma_f32_16x16x32_bf16(
            false, aLo[mt], false, bHi[nt], (short)0, acc[mt][nt], false, false);
      }
  }

  // ---- Epilogue: logits = 2*cross - csq[k] - xsq[n]
  // C/D layout: element v of v8f -> M = v + 8h, N = r
#pragma unroll
  for (int mt = 0; mt < 4; ++mt) {
    const int mrow0 = mBase + wmOff + mt * 16 + 8 * h;
#pragma unroll
    for (int nt = 0; nt < 2; ++nt) {
      const int ncol = nBase + wnOff + nt * 16 + r;
      const float xs = xsq[ncol];
#pragma unroll
      for (int v = 0; v < 8; ++v) {
        const int mrow = mrow0 + v;
        out[(size_t)mrow * N_DIM + ncol] = 2.0f * acc[mt][nt][v] - csq[mrow] - xs;
      }
    }
  }
}

extern "C" void kernel_launch(void* const* d_in, const int* in_sizes, int n_in,
                              void* d_out, int out_size, void* d_ws, size_t ws_size,
                              hipStream_t stream) {
  const float* inputs  = (const float*)d_in[0];   // (8192, 1024)
  const float* centers = (const float*)d_in[1];   // (1024, 1024)
  float* out = (float*)d_out;                     // (1024, 8192)
  float* csq = (float*)d_ws;                      // K_DIM floats
  float* xsq = csq + K_DIM;                       // N_DIM floats

  rownorm_kernel<<<K_DIM / 8, 256, 0, stream>>>(centers, csq);
  rownorm_kernel<<<N_DIM / 8, 256, 0, stream>>>(inputs, xsq);

  dim3 grid(N_DIM / BLK, K_DIM / BLK);  // (64, 8)
  cdist_wmma_kernel<<<grid, 256, 0, stream>>>(inputs, centers, xsq, csq, out);
}
